// LinearLayer_5583457485082
// MI455X (gfx1250) — compile-verified, matrix operand order checked by reading, not run on hardware
//
#include <hip/hip_runtime.h>

typedef __attribute__((ext_vector_type(2)))  float          v2f;
typedef __attribute__((ext_vector_type(4)))  float          v4f;
typedef __attribute__((ext_vector_type(8)))  float          v8f;
typedef __attribute__((ext_vector_type(4)))  unsigned short v4u;
typedef __attribute__((ext_vector_type(8)))  __bf16         v8bf;
typedef __attribute__((ext_vector_type(16))) __bf16         v16bf;

#define DIM_    512
#define LEN_    256
#define BATCH_  16
#define CHAN_   8
#define PAST_   12
#define LOUT_   244          // LEN_ - PAST_
#define NCHUNK_ 64           // N columns per workgroup
#define STRIDE_ 64           // LDS row stride in floats (256*64*4 = 64 KB)
#define NTOT_   (CHAN_ * DIM_)     // 4096
#define AELEMS_ (BATCH_ * LEN_ * DIM_)   // 2,097,152
#define WELEMS_ (NTOT_ * DIM_)           // 2,097,152

// ---------------------------------------------------------------------------
// Prep: split fp32 -> (hi, lo) bf16 planes with RNE rounding.
//   hi = bf16_rne(x); lo = bf16_rne(x - float(hi))   =>  ~16 mantissa bits.
// Thread i handles one float4. i < AELEMS_/4 -> inp, else -> Ws.
// ---------------------------------------------------------------------------
__global__ __launch_bounds__(256)
void split_bf16_kernel(const float* __restrict__ inp, const float* __restrict__ Ws,
                       unsigned short* __restrict__ Ahi, unsigned short* __restrict__ Alo,
                       unsigned short* __restrict__ Whi, unsigned short* __restrict__ Wlo)
{
    const int i = blockIdx.x * blockDim.x + threadIdx.x;   // float4 index
    const int aq = AELEMS_ / 4;
    const float* src;
    unsigned short *hip_, *lop_;
    int j;
    if (i < aq) { src = inp; hip_ = Ahi; lop_ = Alo; j = i; }
    else        { src = Ws;  hip_ = Whi; lop_ = Wlo; j = i - aq; }

    v4f x = ((const v4f*)src)[j];
    v4u h, l;
    #pragma unroll
    for (int e = 0; e < 4; ++e) {
        unsigned u  = __float_as_uint(x[e]);
        unsigned r  = u + 0x7FFFu + ((u >> 16) & 1u);      // RNE to bf16
        unsigned short hb = (unsigned short)(r >> 16);
        float hf    = __uint_as_float((unsigned)hb << 16); // exact
        float lof   = x[e] - hf;
        unsigned ul = __float_as_uint(lof);
        unsigned rl = ul + 0x7FFFu + ((ul >> 16) & 1u);
        h[e] = hb;
        l[e] = (unsigned short)(rl >> 16);
    }
    ((v4u*)hip_)[j] = h;
    ((v4u*)lop_)[j] = l;
}

// Load one 16x32 bf16 fragment: two contiguous 8-element (16B) chunks per lane.
__device__ __forceinline__ v16bf ldfrag(const __bf16* __restrict__ p) {
    v8bf c0 = *(const v8bf*)(p);
    v8bf c1 = *(const v8bf*)(p + 16);
    return __builtin_shufflevector(c0, c1, 0,1,2,3,4,5,6,7,8,9,10,11,12,13,14,15);
}

#define WMMA_BF16(A, B, C) \
    __builtin_amdgcn_wmma_f32_16x16x32_bf16(false, A, false, B, (short)0, C, false, false)

// ---------------------------------------------------------------------------
// Fused GEMM (bf16x3 split) + temporal filter.
// ---------------------------------------------------------------------------
__global__ __launch_bounds__(256)
void fused_bf16x3_kernel(const __bf16* __restrict__ Ahi, const __bf16* __restrict__ Alo,
                         const __bf16* __restrict__ Whi, const __bf16* __restrict__ Wlo,
                         const float* __restrict__ bs, const float* __restrict__ Wt,
                         const float* __restrict__ bt, float* __restrict__ out)
{
    __shared__ float lds[LEN_ * STRIDE_];

    const int chunk  = blockIdx.x;
    const int b      = blockIdx.y;
    const int n_base = chunk * NCHUNK_;

    const int tid  = threadIdx.x;
    const int wave = tid >> 5;
    const int lane = tid & 31;
    const int lm   = lane & 15;            // M (A) / N (B) within tile
    const int lgk  = (lane >> 4) * 8;      // K sub-offset: 0 or 8

    // Per-lane element offsets (bf16 elements, row stride = DIM_).
    const size_t aoff0 = ((size_t)b * LEN_ + (2 * wave    ) * 16 + lm) * DIM_ + lgk;
    const size_t aoff1 = ((size_t)b * LEN_ + (2 * wave + 1) * 16 + lm) * DIM_ + lgk;
    const size_t boff0 = (size_t)(n_base +  0 + lm) * DIM_ + lgk;
    const size_t boff1 = (size_t)(n_base + 16 + lm) * DIM_ + lgk;
    const size_t boff2 = (size_t)(n_base + 32 + lm) * DIM_ + lgk;
    const size_t boff3 = (size_t)(n_base + 48 + lm) * DIM_ + lgk;

    v8f acc[2][4] = {};

    for (int k = 0; k < DIM_; k += 32) {
        v16bf ah0 = ldfrag(Ahi + aoff0 + k);
        v16bf ah1 = ldfrag(Ahi + aoff1 + k);
        v16bf al0 = ldfrag(Alo + aoff0 + k);
        v16bf al1 = ldfrag(Alo + aoff1 + k);
        v16bf bh0 = ldfrag(Whi + boff0 + k);
        v16bf bh1 = ldfrag(Whi + boff1 + k);
        v16bf bh2 = ldfrag(Whi + boff2 + k);
        v16bf bh3 = ldfrag(Whi + boff3 + k);

        // hi*hi
        acc[0][0] = WMMA_BF16(ah0, bh0, acc[0][0]);
        acc[0][1] = WMMA_BF16(ah0, bh1, acc[0][1]);
        acc[0][2] = WMMA_BF16(ah0, bh2, acc[0][2]);
        acc[0][3] = WMMA_BF16(ah0, bh3, acc[0][3]);
        acc[1][0] = WMMA_BF16(ah1, bh0, acc[1][0]);
        acc[1][1] = WMMA_BF16(ah1, bh1, acc[1][1]);
        acc[1][2] = WMMA_BF16(ah1, bh2, acc[1][2]);
        acc[1][3] = WMMA_BF16(ah1, bh3, acc[1][3]);
        // lo*hi
        acc[0][0] = WMMA_BF16(al0, bh0, acc[0][0]);
        acc[0][1] = WMMA_BF16(al0, bh1, acc[0][1]);
        acc[0][2] = WMMA_BF16(al0, bh2, acc[0][2]);
        acc[0][3] = WMMA_BF16(al0, bh3, acc[0][3]);
        acc[1][0] = WMMA_BF16(al1, bh0, acc[1][0]);
        acc[1][1] = WMMA_BF16(al1, bh1, acc[1][1]);
        acc[1][2] = WMMA_BF16(al1, bh2, acc[1][2]);
        acc[1][3] = WMMA_BF16(al1, bh3, acc[1][3]);
        // hi*lo  (B_lo loaded late to cap fragment liveness)
        v16bf bl0 = ldfrag(Wlo + boff0 + k);
        v16bf bl1 = ldfrag(Wlo + boff1 + k);
        v16bf bl2 = ldfrag(Wlo + boff2 + k);
        v16bf bl3 = ldfrag(Wlo + boff3 + k);
        acc[0][0] = WMMA_BF16(ah0, bl0, acc[0][0]);
        acc[0][1] = WMMA_BF16(ah0, bl1, acc[0][1]);
        acc[0][2] = WMMA_BF16(ah0, bl2, acc[0][2]);
        acc[0][3] = WMMA_BF16(ah0, bl3, acc[0][3]);
        acc[1][0] = WMMA_BF16(ah1, bl0, acc[1][0]);
        acc[1][1] = WMMA_BF16(ah1, bl1, acc[1][1]);
        acc[1][2] = WMMA_BF16(ah1, bl2, acc[1][2]);
        acc[1][3] = WMMA_BF16(ah1, bl3, acc[1][3]);
    }

    // Bias + park spat tile in LDS (C/D: VGPR r, lanes0-15 -> M=r; 16-31 -> M=r+8).
    const int mrow = (lane >> 4) * 8;
    #pragma unroll
    for (int j = 0; j < 4; ++j) {
        const float bsv = bs[n_base + j * 16 + lm];
        #pragma unroll
        for (int i = 0; i < 2; ++i) {
            const int Mb = (2 * wave + i) * 16 + mrow;
            #pragma unroll
            for (int r = 0; r < 8; ++r)
                lds[(Mb + r) * STRIDE_ + j * 16 + lm] = acc[i][j][r] + bsv;
        }
    }
    __syncthreads();

    const int c  = n_base / DIM_;
    const int ob = n_base % DIM_;
    float wt[PAST_];
    #pragma unroll
    for (int p = 0; p < PAST_; ++p) wt[p] = Wt[c * PAST_ + p];
    const float btv = bt[c];

    const int nl = tid & 63;
    const int l0 = tid >> 6;
    for (int l = l0; l < LOUT_; l += 4) {
        float s = btv;
        #pragma unroll
        for (int p = 0; p < PAST_; ++p)
            s += wt[p] * lds[(l + p) * STRIDE_ + nl];
        __builtin_nontemporal_store(
            s, &out[(((size_t)b * LOUT_ + l) * CHAN_ + c) * DIM_ + ob + nl]);
    }
}

// ---------------------------------------------------------------------------
// Fallback: pure FP32 WMMA fused kernel (no scratch needed).
// ---------------------------------------------------------------------------
__global__ __launch_bounds__(256)
void fused_f32_kernel(const float* __restrict__ inp, const float* __restrict__ Ws,
                      const float* __restrict__ bs, const float* __restrict__ Wt,
                      const float* __restrict__ bt, float* __restrict__ out)
{
    __shared__ float lds[LEN_ * STRIDE_];

    const int chunk  = blockIdx.x;
    const int b      = blockIdx.y;
    const int n_base = chunk * NCHUNK_;

    const int tid  = threadIdx.x;
    const int wave = tid >> 5;
    const int lane = tid & 31;
    const int lm   = lane & 15;
    const int lk   = (lane >> 4) << 1;

    const float* __restrict__ arow0 = inp + ((size_t)b * LEN_ + (2 * wave    ) * 16 + lm) * DIM_ + lk;
    const float* __restrict__ arow1 = inp + ((size_t)b * LEN_ + (2 * wave + 1) * 16 + lm) * DIM_ + lk;
    const float* __restrict__ brow0 = Ws + (size_t)(n_base +  0 + lm) * DIM_ + lk;
    const float* __restrict__ brow1 = Ws + (size_t)(n_base + 16 + lm) * DIM_ + lk;
    const float* __restrict__ brow2 = Ws + (size_t)(n_base + 32 + lm) * DIM_ + lk;
    const float* __restrict__ brow3 = Ws + (size_t)(n_base + 48 + lm) * DIM_ + lk;

    v8f acc[2][4] = {};

    #pragma unroll 4
    for (int k = 0; k < DIM_; k += 4) {
        v2f a0 = *(const v2f*)(arow0 + k);
        v2f a1 = *(const v2f*)(arow1 + k);
        v2f b0 = *(const v2f*)(brow0 + k);
        v2f b1 = *(const v2f*)(brow1 + k);
        v2f b2 = *(const v2f*)(brow2 + k);
        v2f b3 = *(const v2f*)(brow3 + k);
        acc[0][0] = __builtin_amdgcn_wmma_f32_16x16x4_f32(false, a0, false, b0, (short)0, acc[0][0], false, false);
        acc[0][1] = __builtin_amdgcn_wmma_f32_16x16x4_f32(false, a0, false, b1, (short)0, acc[0][1], false, false);
        acc[0][2] = __builtin_amdgcn_wmma_f32_16x16x4_f32(false, a0, false, b2, (short)0, acc[0][2], false, false);
        acc[0][3] = __builtin_amdgcn_wmma_f32_16x16x4_f32(false, a0, false, b3, (short)0, acc[0][3], false, false);
        acc[1][0] = __builtin_amdgcn_wmma_f32_16x16x4_f32(false, a1, false, b0, (short)0, acc[1][0], false, false);
        acc[1][1] = __builtin_amdgcn_wmma_f32_16x16x4_f32(false, a1, false, b1, (short)0, acc[1][1], false, false);
        acc[1][2] = __builtin_amdgcn_wmma_f32_16x16x4_f32(false, a1, false, b2, (short)0, acc[1][2], false, false);
        acc[1][3] = __builtin_amdgcn_wmma_f32_16x16x4_f32(false, a1, false, b3, (short)0, acc[1][3], false, false);
    }

    const int mrow = (lane >> 4) * 8;
    #pragma unroll
    for (int j = 0; j < 4; ++j) {
        const float bsv = bs[n_base + j * 16 + lm];
        #pragma unroll
        for (int i = 0; i < 2; ++i) {
            const int Mb = (2 * wave + i) * 16 + mrow;
            #pragma unroll
            for (int r = 0; r < 8; ++r)
                lds[(Mb + r) * STRIDE_ + j * 16 + lm] = acc[i][j][r] + bsv;
        }
    }
    __syncthreads();

    const int c  = n_base / DIM_;
    const int ob = n_base % DIM_;
    float wt[PAST_];
    #pragma unroll
    for (int p = 0; p < PAST_; ++p) wt[p] = Wt[c * PAST_ + p];
    const float btv = bt[c];

    const int nl = tid & 63;
    const int l0 = tid >> 6;
    for (int l = l0; l < LOUT_; l += 4) {
        float s = btv;
        #pragma unroll
        for (int p = 0; p < PAST_; ++p)
            s += wt[p] * lds[(l + p) * STRIDE_ + nl];
        __builtin_nontemporal_store(
            s, &out[(((size_t)b * LOUT_ + l) * CHAN_ + c) * DIM_ + ob + nl]);
    }
}

extern "C" void kernel_launch(void* const* d_in, const int* in_sizes, int n_in,
                              void* d_out, int out_size, void* d_ws, size_t ws_size,
                              hipStream_t stream) {
    const float* inp = (const float*)d_in[0];
    const float* Ws  = (const float*)d_in[1];
    const float* bs  = (const float*)d_in[2];
    const float* Wt  = (const float*)d_in[3];
    const float* bt  = (const float*)d_in[4];
    float* out = (float*)d_out;

    const size_t plane = (size_t)AELEMS_ * sizeof(unsigned short);  // 4 MB
    const size_t need  = 4 * plane;                                 // 16 MB

    dim3 grid(NTOT_ / NCHUNK_, BATCH_);   // 64 x 16

    if (ws_size >= need) {
        unsigned short* Ahi = (unsigned short*)d_ws;
        unsigned short* Alo = Ahi + AELEMS_;
        unsigned short* Whi = Alo + AELEMS_;
        unsigned short* Wlo = Whi + WELEMS_;

        const int n4 = (AELEMS_ + WELEMS_) / 4;                     // 1,048,576
        split_bf16_kernel<<<n4 / 256, 256, 0, stream>>>(inp, Ws, Ahi, Alo, Whi, Wlo);

        fused_bf16x3_kernel<<<grid, 256, 0, stream>>>(
            (const __bf16*)Ahi, (const __bf16*)Alo,
            (const __bf16*)Whi, (const __bf16*)Wlo, bs, Wt, bt, out);
    } else {
        fused_f32_kernel<<<grid, 256, 0, stream>>>(inp, Ws, bs, Wt, bt, out);
    }
}
